// HODERG_5652176961834
// MI455X (gfx1250) — compile-verified, matrix operand order checked by reading, not run on hardware
//
#include <hip/hip_runtime.h>
#include <hip/hip_bf16.h>

typedef unsigned int  u32;
typedef unsigned short u16;

typedef __attribute__((ext_vector_type(16))) __bf16 v16bf;
typedef __attribute__((ext_vector_type(8)))  float  v8f;
typedef __attribute__((ext_vector_type(4)))  unsigned int v4u;
typedef __attribute__((ext_vector_type(8)))  int v8i;
typedef __attribute__((ext_vector_type(4)))  int v4i;

#define USE_TDM 1

union Frag { v16bf v; u32 u[8]; };

// ---- helpers -------------------------------------------------------------

__device__ __host__ inline u16 f2bfu(float f) {
    u32 u = __builtin_bit_cast(u32, f);
    u32 r = u + 0x7FFFu + ((u >> 16) & 1u);   // round-to-nearest-even
    return (u16)(r >> 16);
}

// Packed-pair K index inside a 32-wide K chunk for 16-bit A/B fragments:
// lanes 0-15 hold K {0..7,16..23}, lanes 16-31 hold K {8..15,24..31};
// VGPR j packs (K, K+1).
__device__ inline int kkOf(int j, int lhalf) {
    return ((j & 4) << 2) + (lhalf << 3) + ((j & 3) << 1);
}

__device__ inline v8f wmma_bf16(v16bf a, v16bf b, v8f c) {
    return __builtin_amdgcn_wmma_f32_16x16x32_bf16(
        false, a, false, b, (short)0, c, false, false);
}

// ds_swizzle xor butterfly within 16-lane groups (immediate pattern, no
// address VALU): offset = xor_mask<<10 | and_mask(0x1f)
#define SWIZ(v, imm) __builtin_bit_cast(float, \
    __builtin_amdgcn_ds_swizzle(__builtin_bit_cast(int, (v)), (imm)))

__device__ inline float redmax16(float v) {
    v = fmaxf(v, SWIZ(v, 0x041f));
    v = fmaxf(v, SWIZ(v, 0x081f));
    v = fmaxf(v, SWIZ(v, 0x101f));
    v = fmaxf(v, SWIZ(v, 0x201f));
    return v;
}
__device__ inline float redsum16(float v) {
    v += SWIZ(v, 0x041f);
    v += SWIZ(v, 0x081f);
    v += SWIZ(v, 0x101f);
    v += SWIZ(v, 0x201f);
    return v;
}

// ---- fp32 -> bf16 convert ------------------------------------------------

__global__ void cvt_kernel(const float* __restrict__ in, u16* __restrict__ out, int n) {
    int i = blockIdx.x * 256 + threadIdx.x;
    if (i < n) out[i] = f2bfu(in[i]);
}

// ---- LayerNorm (one block per row, D = 512) ------------------------------

__global__ __launch_bounds__(256)
void ln_kernel(const float* __restrict__ x, const float* __restrict__ g,
               const float* __restrict__ bt, u16* __restrict__ ybf,
               float* __restrict__ yf) {
    const int D = 512;
    int row = blockIdx.x;
    int t = threadIdx.x;
    const float* xr = x + (size_t)row * D;
    float v0 = xr[t], v1 = xr[t + 256];
    float s = v0 + v1, ss = v0 * v0 + v1 * v1;
    #pragma unroll
    for (int off = 16; off >= 1; off >>= 1) {
        s  += __shfl_xor(s,  off, 32);
        ss += __shfl_xor(ss, off, 32);
    }
    __shared__ float sh[16];
    int wid = t >> 5, ln = t & 31;
    if (ln == 0) { sh[wid] = s; sh[wid + 8] = ss; }
    __syncthreads();
    if (t == 0) {
        float S = 0.f, SS = 0.f;
        #pragma unroll
        for (int i = 0; i < 8; ++i) { S += sh[i]; SS += sh[i + 8]; }
        float mean = S / D;
        float var  = SS / D - mean * mean;
        sh[0] = mean;
        sh[1] = rsqrtf(var + 1e-5f);
    }
    __syncthreads();
    float mean = sh[0], rs = sh[1];
    float y0 = (v0 - mean) * rs * g[t]       + bt[t];
    float y1 = (v1 - mean) * rs * g[t + 256] + bt[t + 256];
    size_t base = (size_t)row * D;
    ybf[base + t]       = f2bfu(y0);
    ybf[base + t + 256] = f2bfu(y1);
    if (yf) { yf[base + t] = y0; yf[base + t + 256] = y1; }
}

// ---- bf16 WMMA GEMM: C[M,N] = A[M,K] * W[K,N] (+epilogue) ----------------
// Block: 256 threads = 8 waves; macro-tile 128x64; wave tile 32x32 (2x2 WMMA).
// Double-buffered: next global tile prefetched into VGPRs during compute.
// MODE 0: out bf16 = (acc + bias) * scale              (q/k/v projections)
// MODE 1: out f32  = aux1 * sigmoid(acc + bias)        (GLU gate -> u)
// MODE 2: out f32  = acc + bias + aux1 [+ aux2]        (Wo resid / W2 resid)
// MODE 3: out bf16 = gelu_exact(acc + bias)            (FFN hidden)

template <int MODE>
__global__ __launch_bounds__(256)
void gemm_bf16(const u16* __restrict__ A, const u16* __restrict__ W,
               const float* __restrict__ bias, const float* __restrict__ aux1,
               const float* __restrict__ aux2, void* __restrict__ outp,
               int M, int K, int Nn, float scale) {
    __shared__ __align__(16) u16 As[128 * 40];   // [m][k], stride 40 (16B rows)
    __shared__ __align__(16) u16 BsT[64 * 34];   // [n][k], stride 34 (odd dword)

    int tid = threadIdx.x;
    int lane = tid & 31, wave = tid >> 5;
    int warpM = wave & 3, warpN = wave >> 2;
    int mBase = blockIdx.x * 128, nBase = blockIdx.y * 64;
    int lhalf = lane >> 4, lr = lane & 15;

    v8f acc[2][2] = {};

    int arow = tid >> 2, acol = (tid & 3) * 8;   // A coop load: 64 rows/pass x 32
    int brow = tid >> 3, bcol = (tid & 7) * 8;   // B coop load: 32 x 64

    uint4 aReg0, aReg1;
    union { uint4 q; u16 s[8]; } bReg;
    auto gload = [&](int kb) {
        aReg0  = *(const uint4*)(A + (size_t)(mBase + arow)      * K + kb + acol);
        aReg1  = *(const uint4*)(A + (size_t)(mBase + arow + 64) * K + kb + acol);
        bReg.q = *(const uint4*)(W + (size_t)(kb + brow) * Nn + nBase + bcol);
    };
    gload(0);

    for (int kb = 0; kb < K; kb += 32) {
        // commit prefetched tile to LDS
        *(uint4*)&As[arow * 40 + acol]        = aReg0;
        *(uint4*)&As[(arow + 64) * 40 + acol] = aReg1;
        #pragma unroll
        for (int i = 0; i < 8; ++i) BsT[(bcol + i) * 34 + brow] = bReg.s[i];
        __syncthreads();

        if (kb + 32 < K) gload(kb + 32);   // overlap next global fetch with WMMAs

        Frag af[2], bfr[2];
        #pragma unroll
        for (int m = 0; m < 2; ++m)
            #pragma unroll
            for (int j = 0; j < 8; ++j)
                af[m].u[j] = *(const u32*)&As[(warpM * 32 + m * 16 + lr) * 40 + kkOf(j, lhalf)];
        #pragma unroll
        for (int n = 0; n < 2; ++n)
            #pragma unroll
            for (int j = 0; j < 8; ++j)
                bfr[n].u[j] = *(const u32*)&BsT[(warpN * 32 + n * 16 + lr) * 34 + kkOf(j, lhalf)];
        #pragma unroll
        for (int m = 0; m < 2; ++m)
            #pragma unroll
            for (int n = 0; n < 2; ++n)
                acc[m][n] = wmma_bf16(af[m].v, bfr[n].v, acc[m][n]);
        __syncthreads();
    }

    // Epilogue: C/D layout -> lane 0-15 holds M=r, lane 16-31 holds M=r+8.
    #pragma unroll
    for (int m = 0; m < 2; ++m)
        #pragma unroll
        for (int n = 0; n < 2; ++n)
            #pragma unroll
            for (int r = 0; r < 8; ++r) {
                int row = mBase + warpM * 32 + m * 16 + r + lhalf * 8;
                int col = nBase + warpN * 32 + n * 16 + lr;
                size_t idx = (size_t)row * Nn + col;
                float v = acc[m][n][r] + bias[col];
                if constexpr (MODE == 0) {
                    ((u16*)outp)[idx] = f2bfu(v * scale);
                } else if constexpr (MODE == 1) {
                    ((float*)outp)[idx] =
                        aux1[idx] * __builtin_amdgcn_rcpf(1.f + __expf(-v));
                } else if constexpr (MODE == 2) {
                    float o = v + aux1[idx];
                    if (aux2) o += aux2[idx];
                    ((float*)outp)[idx] = o;
                } else {
                    float ge = 0.5f * v * (1.f + erff(v * 0.70710678118654752f));
                    ((u16*)outp)[idx] = f2bfu(ge);
                }
            }
}

// ---- Flash attention: scores = qk^T + bias, mask, softmax, *v ------------
// Grid (N/64, H, B); block 128 threads = 4 waves, 16 q-rows per wave.
// K chunk staged via the Tensor Data Mover (2D tile w/ LDS padding that
// reproduces the 72-ushort row stride); V staged manually (transposed).

__global__ __launch_bounds__(128)
void attn_kernel(const u16* __restrict__ qb, const u16* __restrict__ kbuf,
                 const u16* __restrict__ vbuf, const float* __restrict__ bias,
                 const unsigned char* __restrict__ mask, u16* __restrict__ attnb) {
    constexpr int Nn = 2048, Hh = 8, Dd = 512, HD = 64;
    __shared__ __align__(16) u16 kT[32 * 72];     // [key][d], stride 72 (=128B row + 16B pad)
    __shared__ __align__(16) u16 vT[64 * 34];     // [d][key], stride 34 (transposed)
    __shared__ __align__(16) u16 pS[4 * 16 * 34]; // per-wave P staging [16][32+pad]

    int t = threadIdx.x, lane = t & 31, wave = t >> 5;
    int lhalf = lane >> 4, lr = lane & 15;
    int b = blockIdx.z, h = blockIdx.y;
    int qBase = blockIdx.x * 64 + wave * 16;

    // q A-fragments (16x64 = two 16x32 chunks), read once from global
    Frag qf[2];
    const u16* qp = qb + ((size_t)(b * Nn + qBase + lr)) * Dd + h * HD;
    #pragma unroll
    for (int c = 0; c < 2; ++c)
        #pragma unroll
        for (int j = 0; j < 8; ++j)
            qf[c].u[j] = *(const u32*)(qp + c * 32 + kkOf(j, lhalf));

#if USE_TDM
    // Tensor DMA descriptor (constant parts). D# group1 per ISA 8.4:
    //   data_size=1 (2B) @bit16, pad_enable @bit20, pad_interval=4 (32 dwords)
    //   @bits24:22, pad_amount=3 (4 dwords) @bits31:25,
    //   tensor_dim0=512, tensor_dim1=2048, tile_dim0=64, tile_dim1=32,
    //   tensor_dim0_stride=512.
    const v8i tdmG1 = { (int)((1u << 16) | (1u << 20) | (4u << 22) | (3u << 25)),
                        (int)(512u  << 16),
                        (int)(2048u << 16),
                        (int)(64u   << 16),
                        32, 512, 0, 0 };
    const v4i tdmG2 = { 0, 0, 0, 0 };
    const v4i tdmG3 = { 0, 0, 0, 0 };
    const v8i tdmG4 = { 0, 0, 0, 0, 0, 0, 0, 0 };
    u32 ldsAddr = (u32)(unsigned long long)(&kT[0]);
    const u16* kChunkBase = kbuf + ((size_t)(b * Nn)) * Dd + h * HD;
#endif

    float mrow[8], lrow[8];
    #pragma unroll
    for (int r = 0; r < 8; ++r) { mrow[r] = -1e30f; lrow[r] = 0.f; }
    v8f o[4] = {};

    for (int keyBase = 0; keyBase < Nn; keyBase += 32) {
        __syncthreads();
#if USE_TDM
        if (wave == 0) {
            unsigned long long ga =
                (unsigned long long)(const void*)(kChunkBase + (size_t)keyBase * Dd);
            // group0: count=1 | lds_addr | global_addr | type=2 (bits 127:126)
            v4u g0 = { 1u, ldsAddr, (u32)ga,
                       (u32)((ga >> 32) & 0x01FFFFFFu) | 0x80000000u };
            __builtin_amdgcn_tensor_load_to_lds(g0, tdmG1, tdmG2, tdmG3, tdmG4, 0);
        }
#else
        {
            int krow = t >> 2, kcol = (t & 3) * 16;
            const uint4* ks = (const uint4*)(kbuf + ((size_t)(b * Nn + keyBase + krow)) * Dd + h * HD + kcol);
            *(uint4*)&kT[krow * 72 + kcol]     = ks[0];
            *(uint4*)&kT[krow * 72 + kcol + 8] = ks[1];
        }
#endif
        {   // V chunk, transposed into LDS
            int krow = t >> 2, kcol = (t & 3) * 16;
            union { uint4 q[2]; u16 s[16]; } tv;
            const uint4* vs = (const uint4*)(vbuf + ((size_t)(b * Nn + keyBase + krow)) * Dd + h * HD + kcol);
            tv.q[0] = vs[0]; tv.q[1] = vs[1];
            #pragma unroll
            for (int i = 0; i < 16; ++i) vT[(kcol + i) * 34 + krow] = tv.s[i];
        }
#if USE_TDM
        if (wave == 0) __builtin_amdgcn_s_wait_tensorcnt(0);
#endif
        __syncthreads();

        // S = q * k^T  (two 16x16 tiles over 32 keys, K-dim 64 -> 2 chained WMMAs each)
        Frag bk[2][2];
        #pragma unroll
        for (int s = 0; s < 2; ++s)
            #pragma unroll
            for (int c = 0; c < 2; ++c)
                #pragma unroll
                for (int j = 0; j < 8; ++j)
                    bk[s][c].u[j] = *(const u32*)&kT[(s * 16 + lr) * 72 + c * 32 + kkOf(j, lhalf)];
        v8f S0 = {}, S1 = {};
        S0 = wmma_bf16(qf[0].v, bk[0][0].v, S0);
        S0 = wmma_bf16(qf[1].v, bk[0][1].v, S0);
        S1 = wmma_bf16(qf[0].v, bk[1][0].v, S1);
        S1 = wmma_bf16(qf[1].v, bk[1][1].v, S1);

        // bias + mask + online softmax (row halves match C-layout lane halves)
        #pragma unroll
        for (int r = 0; r < 8; ++r) {
            int qrow = qBase + r + lhalf * 8;
            size_t rowk = ((size_t)(b * Nn + qrow)) * Nn + keyBase + lr;
            if (keyBase + 64 < Nn) {   // keep the 268MB bias stream ahead of WMMA
                __builtin_prefetch(&bias[(rowk + 64) * Hh + h], 0, 0);
                __builtin_prefetch(&bias[(rowk + 80) * Hh + h], 0, 0);
            }
            float s0 = S0[r] + bias[rowk * Hh + h];
            float s1 = S1[r] + bias[(rowk + 16) * Hh + h];
            if (mask[rowk])      s0 = -1e30f;
            if (mask[rowk + 16]) s1 = -1e30f;
            float mx = redmax16(fmaxf(s0, s1));
            float mnew = fmaxf(mrow[r], mx);
            float sc = (mrow[r] < -9e29f) ? 0.f : __expf(mrow[r] - mnew);
            float p0 = __expf(s0 - mnew), p1 = __expf(s1 - mnew);
            float ps = redsum16(p0 + p1);
            lrow[r] = lrow[r] * sc + ps;
            mrow[r] = mnew;
            #pragma unroll
            for (int n = 0; n < 4; ++n) o[n][r] *= sc;
            int prow = r + lhalf * 8;
            pS[wave * 544 + prow * 34 + lr]      = f2bfu(p0);
            pS[wave * 544 + prow * 34 + lr + 16] = f2bfu(p1);
        }

        // o += P * V  (P via LDS C->A layout transpose, V transposed in LDS)
        Frag pa;
        #pragma unroll
        for (int j = 0; j < 8; ++j)
            pa.u[j] = *(const u32*)&pS[wave * 544 + lr * 34 + kkOf(j, lhalf)];
        #pragma unroll
        for (int n = 0; n < 4; ++n) {
            Frag bv;
            #pragma unroll
            for (int j = 0; j < 8; ++j)
                bv.u[j] = *(const u32*)&vT[(n * 16 + lr) * 34 + kkOf(j, lhalf)];
            o[n] = wmma_bf16(pa.v, bv.v, o[n]);
        }
    }

    #pragma unroll
    for (int r = 0; r < 8; ++r) {
        float inv = __builtin_amdgcn_rcpf(lrow[r]);
        int qrow = qBase + r + lhalf * 8;
        #pragma unroll
        for (int n = 0; n < 4; ++n)
            attnb[((size_t)(b * Nn + qrow)) * Dd + h * HD + n * 16 + lr] =
                f2bfu(o[n][r] * inv);
    }
}

// ---- host orchestration --------------------------------------------------

extern "C" void kernel_launch(void* const* d_in, const int* in_sizes, int n_in,
                              void* d_out, int out_size, void* d_ws, size_t ws_size,
                              hipStream_t stream) {
    constexpr int Bb = 2, Nn = 2048, Dd = 512, Hh = 8, Ff = 2048;
    const int M = Bb * Nn;

    const float* x         = (const float*)d_in[0];
    const float* attn_bias = (const float*)d_in[1];
    const unsigned char* attn_mask = (const unsigned char*)d_in[2];
    const float* Wq = (const float*)d_in[3];  const float* bq = (const float*)d_in[4];
    const float* Wk = (const float*)d_in[5];  const float* bk = (const float*)d_in[6];
    const float* Wv = (const float*)d_in[7];  const float* bv = (const float*)d_in[8];
    const float* Wg = (const float*)d_in[9];  const float* bg = (const float*)d_in[10];
    const float* Wo = (const float*)d_in[11]; const float* bo = (const float*)d_in[12];
    const float* ln1g = (const float*)d_in[13]; const float* ln1b = (const float*)d_in[14];
    const float* ln2g = (const float*)d_in[15]; const float* ln2b = (const float*)d_in[16];
    const float* W1 = (const float*)d_in[17]; const float* b1 = (const float*)d_in[18];
    const float* W2 = (const float*)d_in[19]; const float* b2 = (const float*)d_in[20];

    char* wp = (char*)d_ws;
    auto alloc = [&](size_t bytes) -> void* {
        void* p = (void*)wp;
        wp += (bytes + 255) & ~(size_t)255;
        return p;
    };
    u16*   yb    = (u16*)  alloc((size_t)M * Dd * 2);   // LN1 out, bf16
    float* yf    = (float*)alloc((size_t)M * Dd * 4);   // LN1 out, fp32 (for GLU)
    u16*   qbf   = (u16*)  alloc((size_t)M * Dd * 2);
    u16*   kbf   = (u16*)  alloc((size_t)M * Dd * 2);
    u16*   vbf   = (u16*)  alloc((size_t)M * Dd * 2);
    float* ubuf  = (float*)alloc((size_t)M * Dd * 4);   // GLU branch
    u16*   attnb = (u16*)  alloc((size_t)M * Dd * 2);   // attention out, bf16
    float* x2    = (float*)alloc((size_t)M * Dd * 4);   // after first residual
    u16*   y2b   = (u16*)  alloc((size_t)M * Dd * 2);   // LN2 out, bf16
    u16*   hb    = (u16*)  alloc((size_t)M * Ff * 2);   // FFN hidden, bf16
    u16*   Wqb = (u16*)alloc((size_t)Dd * Dd * 2);
    u16*   Wkb = (u16*)alloc((size_t)Dd * Dd * 2);
    u16*   Wvb = (u16*)alloc((size_t)Dd * Dd * 2);
    u16*   Wgb = (u16*)alloc((size_t)Dd * Dd * 2);
    u16*   Wob = (u16*)alloc((size_t)Dd * Dd * 2);
    u16*   W1b = (u16*)alloc((size_t)Dd * Ff * 2);
    u16*   W2b = (u16*)alloc((size_t)Ff * Dd * 2);

    // weights -> bf16
    const int nDD = Dd * Dd, nDF = Dd * Ff;
    cvt_kernel<<<(nDD + 255) / 256, 256, 0, stream>>>(Wq, Wqb, nDD);
    cvt_kernel<<<(nDD + 255) / 256, 256, 0, stream>>>(Wk, Wkb, nDD);
    cvt_kernel<<<(nDD + 255) / 256, 256, 0, stream>>>(Wv, Wvb, nDD);
    cvt_kernel<<<(nDD + 255) / 256, 256, 0, stream>>>(Wg, Wgb, nDD);
    cvt_kernel<<<(nDD + 255) / 256, 256, 0, stream>>>(Wo, Wob, nDD);
    cvt_kernel<<<(nDF + 255) / 256, 256, 0, stream>>>(W1, W1b, nDF);
    cvt_kernel<<<(nDF + 255) / 256, 256, 0, stream>>>(W2, W2b, nDF);

    // LN1
    ln_kernel<<<M, 256, 0, stream>>>(x, ln1g, ln1b, yb, yf);

    // projections
    dim3 gD(M / 128, Dd / 64);
    gemm_bf16<0><<<gD, 256, 0, stream>>>(yb, Wqb, bq, nullptr, nullptr, qbf, M, Dd, Dd, 0.125f);
    gemm_bf16<0><<<gD, 256, 0, stream>>>(yb, Wkb, bk, nullptr, nullptr, kbf, M, Dd, Dd, 1.0f);
    gemm_bf16<0><<<gD, 256, 0, stream>>>(yb, Wvb, bv, nullptr, nullptr, vbf, M, Dd, Dd, 1.0f);
    gemm_bf16<1><<<gD, 256, 0, stream>>>(yb, Wgb, bg, yf, nullptr, ubuf, M, Dd, Dd, 1.0f);

    // attention
    attn_kernel<<<dim3(Nn / 64, Hh, Bb), 128, 0, stream>>>(qbf, kbf, vbf, attn_bias, attn_mask, attnb);

    // Wo + u + residual -> x2
    gemm_bf16<2><<<gD, 256, 0, stream>>>(attnb, Wob, bo, ubuf, x, x2, M, Dd, Dd, 1.0f);

    // LN2 + FFN
    ln_kernel<<<M, 256, 0, stream>>>(x2, ln2g, ln2b, y2b, nullptr);
    dim3 gF(M / 128, Ff / 64);
    gemm_bf16<3><<<gF, 256, 0, stream>>>(y2b, W1b, b1, nullptr, nullptr, hb, M, Dd, Ff, 1.0f);
    gemm_bf16<2><<<gD, 256, 0, stream>>>(hb, W2b, b2, x2, nullptr, (float*)d_out, M, Ff, Dd, 1.0f);
}
